// DRNN_75196287418412
// MI455X (gfx1250) — compile-verified, matrix operand order checked by reading, not run on hardware
//
#include <hip/hip_runtime.h>
#include <hip/hip_bf16.h>

// ---------------------------------------------------------------------------
// DialogueRNN on MI455X (gfx1250):
//  - bf16 weights resident in 192MB L2 (~50MB), fp32 accumulation via
//    v_wmma_f32_16x16x32_bf16 (wave32 WMMA).
//  - Per-step skinny GEMMs (M=64/128, N=3072, K=1024/2048): gi+gh batched in
//    one launch (blockIdx.z), LDS double-buffered panels staged with
//    GLOBAL_LOAD_ASYNC_TO_LDS_B128 (ASYNCcnt) when the toolchain exposes it.
//  - Attention scores cached per step -> O(t*DG) softmax/context.
// ---------------------------------------------------------------------------

typedef __bf16 v16bf __attribute__((ext_vector_type(16)));
typedef float  v8f   __attribute__((ext_vector_type(8)));
typedef unsigned int u32x4 __attribute__((ext_vector_type(4)));

// Async-to-LDS builtin takes pointers to 16B int vectors (per hipcc diag):
typedef int v4i_t __attribute__((vector_size(16)));
typedef __attribute__((address_space(1))) v4i_t as1_v4i;
typedef __attribute__((address_space(3))) v4i_t as3_v4i;

union FragBF { v16bf v; u32x4 q[2]; };

#if __has_builtin(__builtin_amdgcn_global_load_async_to_lds_b128) && \
    __has_builtin(__builtin_amdgcn_s_wait_asynccnt)
#define ASYNC_LDS 1
#endif

#define T_STEPS 128
#define BATCH   64
#define SPK     2
#define DHID    1024
#define N3      3072   // 3*DHID gate columns

__device__ __forceinline__ unsigned short f2bf(float f) {
    unsigned int u = __float_as_uint(f);
    unsigned int r = u + 0x7fffu + ((u >> 16) & 1u);   // round-nearest-even
    return (unsigned short)(r >> 16);
}
__device__ __forceinline__ float sigmoidf(float x) {
    return 1.0f / (1.0f + __expf(-x));
}

// 16B global -> LDS copy: async DMA path (ASYNCcnt) or sync fallback.
__device__ __forceinline__ void copy16_g2l(const unsigned short* g,
                                           unsigned short* l) {
#ifdef ASYNC_LDS
    __builtin_amdgcn_global_load_async_to_lds_b128(
        (as1_v4i*)g, (as3_v4i*)l, 0, 0);
#else
    *(u32x4*)l = *(const u32x4*)g;
#endif
}
__device__ __forceinline__ void async_wait_prev() {  // oldest stage complete
#ifdef ASYNC_LDS
    __builtin_amdgcn_s_wait_asynccnt(3);
#endif
}
__device__ __forceinline__ void async_wait_all() {
#ifdef ASYNC_LDS
    __builtin_amdgcn_s_wait_asynccnt(0);
#endif
}

// ---------------------------------------------------------------------------
// Utility kernels
// ---------------------------------------------------------------------------
__global__ void cvt_f32_bf16_kernel(const float* __restrict__ src,
                                    unsigned short* __restrict__ dst, size_t n) {
    size_t i = (size_t)blockIdx.x * blockDim.x + threadIdx.x;
    size_t stride = (size_t)gridDim.x * blockDim.x;
    for (; i < n; i += stride) dst[i] = f2bf(src[i]);
}

__global__ void zero_u32_kernel(unsigned int* __restrict__ p, size_t n) {
    size_t i = (size_t)blockIdx.x * blockDim.x + threadIdx.x;
    size_t stride = (size_t)gridDim.x * blockDim.x;
    for (; i < n; i += stride) p[i] = 0u;
}

__global__ void spk_idx_kernel(const float* __restrict__ speakers,
                               int* __restrict__ spk, int n) {
    int i = blockIdx.x * blockDim.x + threadIdx.x;
    if (i >= n) return;
    int best = 0;
    #pragma unroll
    for (int s = 0; s < SPK; ++s)
        if (speakers[(size_t)i * SPK + s] > 0.5f) best = s;
    spk[i] = best;
}

// ---------------------------------------------------------------------------
// Dual WMMA GEMM: one launch computes two gate projections of a GRU step.
//   z == 0          : C0[64  x 3072] = A0[64 x K0] * W0[3072 x K0]^T
//   z >= 1          : C1 rows (z-1)*64..  = A1 * W1^T   (K1), M1 = 64*(zTot-1)
// Workgroup = 8 wave32s -> 64(M) x 32(N) tile; double-buffered LDS panels
// staged via async-to-LDS; 2 WMMAs per 64-wide K chunk per wave.
// ---------------------------------------------------------------------------
__global__ __launch_bounds__(256)
void gemm_dual_bf16_wmma_kernel(const unsigned short* __restrict__ A0,
                                const unsigned short* __restrict__ W0,
                                float* __restrict__ C0, int K0,
                                const unsigned short* __restrict__ A1,
                                const unsigned short* __restrict__ W1,
                                float* __restrict__ C1, int K1) {
    __shared__ unsigned short lA[2][64][72];   // 144B row stride (16B aligned)
    __shared__ unsigned short lB[2][32][72];

    const int z = blockIdx.z;
    const unsigned short* A;
    const unsigned short* W;
    float* C;
    int K, m_base;
    if (z == 0) { A = A0; W = W0; C = C0; K = K0; m_base = 0; }
    else        { A = A1; W = W1; C = C1; K = K1; m_base = (z - 1) * 64; }

    const int tid   = threadIdx.x;
    const int lane  = tid & 31;
    const int wave  = tid >> 5;
    const int mt    = wave & 3;      // 4 M-tiles of 16
    const int nt    = wave >> 2;     // 2 N-tiles of 16
    const int n_base = blockIdx.x * 32;

    const int mr    = lane & 15;
    const int khalf = lane >> 4;     // 16-bit A/B layout: lanes 16-31 hold K+8
    const int kofs  = khalf * 8;

    // one stage = 3 async instructions per wave (2 A-chunks + 1 B-chunk)
    auto stage = [&](int c, int bufi) {
        const int k0 = c << 6;
        int q = tid;
        #pragma unroll
        for (int it = 0; it < 2; ++it, q += 256) {
            int row = q >> 3, cc = (q & 7) * 8;
            copy16_g2l(A + (size_t)(m_base + row) * K + k0 + cc,
                       &lA[bufi][row][cc]);
        }
        int row = tid >> 3, cc = (tid & 7) * 8;
        copy16_g2l(W + (size_t)(n_base + row) * K + k0 + cc,
                   &lB[bufi][row][cc]);
    };

    v8f acc = {};
    const int nch = K >> 6;          // 64-wide K chunks (>= 16 here)

    stage(0, 0);
    if (nch > 1) stage(1, 1);

    for (int c = 0; c < nch; ++c) {
        const int cur = c & 1;
        if (c + 1 < nch) async_wait_prev(); else async_wait_all();
        __syncthreads();

        #pragma unroll
        for (int ks = 0; ks < 64; ks += 32) {
            FragBF a, b;
            const int arow = mt * 16 + mr;
            const int brow = nt * 16 + mr;
            a.q[0] = *(const u32x4*)&lA[cur][arow][ks + kofs];
            a.q[1] = *(const u32x4*)&lA[cur][arow][ks + kofs + 16];
            b.q[0] = *(const u32x4*)&lB[cur][brow][ks + kofs];
            b.q[1] = *(const u32x4*)&lB[cur][brow][ks + kofs + 16];
            acc = __builtin_amdgcn_wmma_f32_16x16x32_bf16(
                false, a.v, false, b.v, (short)0, acc, false, false);
        }
        __syncthreads();

        if (c + 2 < nch) stage(c + 2, cur);
        if (c + 4 < nch) {           // keep weight panel warm (global_prefetch_b8)
            int row = tid >> 3;
            __builtin_prefetch(W + (size_t)(n_base + row) * K +
                               ((c + 4) << 6) + (tid & 7) * 8, 0, 2);
        }
    }

    // C/D layout: lanes 0-15 -> M=r, lanes 16-31 -> M=8+r; N = lane&15
    const int n = n_base + nt * 16 + (lane & 15);
    #pragma unroll
    for (int r = 0; r < 8; ++r) {
        int m = m_base + mt * 16 + khalf * 8 + r;
        C[(size_t)m * N3 + n] = acc[r];
    }
}

// ---------------------------------------------------------------------------
// Concat input for global GRU: xg = [feat_t , parties[b, spk_b, :]] (bf16)
// ---------------------------------------------------------------------------
__global__ void build_xg_kernel(const unsigned short* __restrict__ featT,
                                const float* __restrict__ parties,
                                const int* __restrict__ spk,
                                unsigned short* __restrict__ xg) {
    int i = blockIdx.x * blockDim.x + threadIdx.x;   // BATCH*DHID
    if (i >= BATCH * DHID) return;
    int b = i >> 10, d = i & 1023;
    xg[(size_t)b * 2048 + d] = featT[i];
    int sp = spk[b];
    xg[(size_t)b * 2048 + 1024 + d] =
        f2bf(parties[((size_t)(b * SPK + sp) << 10) + d]);
}

// ---------------------------------------------------------------------------
// Fused GRU gate math (torch GRUCell order r,z,n); writes fp32 + bf16 state.
// ---------------------------------------------------------------------------
__global__ void gate_gru_kernel(const float* __restrict__ gi,
                                const float* __restrict__ gh,
                                const float* __restrict__ h,
                                const float* __restrict__ bih,
                                const float* __restrict__ bhh,
                                float* __restrict__ out,
                                unsigned short* __restrict__ out_bf) {
    int i = blockIdx.x * blockDim.x + threadIdx.x;   // BATCH*DHID
    if (i >= BATCH * DHID) return;
    int b = i >> 10, d = i & 1023;
    size_t r3 = (size_t)b * N3;
    float ir = gi[r3 + d]          + bih[d];
    float iz = gi[r3 + 1024 + d]   + bih[1024 + d];
    float in = gi[r3 + 2048 + d]   + bih[2048 + d];
    float hr = gh[r3 + d]          + bhh[d];
    float hz = gh[r3 + 1024 + d]   + bhh[1024 + d];
    float hn = gh[r3 + 2048 + d]   + bhh[2048 + d];
    float rg = sigmoidf(ir + hr);
    float zg = sigmoidf(iz + hz);
    float ng = tanhf(in + rg * hn);
    float o  = (1.0f - zg) * ng + zg * h[i];
    out[i] = o;
    out_bf[i] = f2bf(o);
}

// Party GRU gates + masked scatter to the speaker slot; emits e_in (bf16).
__global__ void gate_party_kernel(const float* __restrict__ gi,   // B x 3072
                                  const float* __restrict__ gh,   // (B*S) x 3072
                                  float* __restrict__ parties,    // B*S*D (in/out)
                                  const float* __restrict__ bih,
                                  const float* __restrict__ bhh,
                                  const int* __restrict__ spk,
                                  unsigned short* __restrict__ parties_bf,
                                  unsigned short* __restrict__ ein_bf) {
    int i = blockIdx.x * blockDim.x + threadIdx.x;   // BATCH*SPK*DHID
    if (i >= BATCH * SPK * DHID) return;
    int b = i >> 11, s = (i >> 10) & 1, d = i & 1023;
    size_t gi3 = (size_t)b * N3;
    size_t gh3 = (size_t)(b * SPK + s) * N3;
    float ir = gi[gi3 + d]        + bih[d];
    float iz = gi[gi3 + 1024 + d] + bih[1024 + d];
    float in = gi[gi3 + 2048 + d] + bih[2048 + d];
    float hr = gh[gh3 + d]        + bhh[d];
    float hz = gh[gh3 + 1024 + d] + bhh[1024 + d];
    float hn = gh[gh3 + 2048 + d] + bhh[2048 + d];
    float hv = parties[i];
    float rg = sigmoidf(ir + hr);
    float zg = sigmoidf(iz + hz);
    float ng = tanhf(in + rg * hn);
    float q  = (1.0f - zg) * ng + zg * hv;
    int sp = spk[b];
    float nv = (s == sp) ? q : hv;
    parties[i] = nv;
    parties_bf[i] = f2bf(nv);
    if (s == sp) ein_bf[((size_t)b << 10) + d] = f2bf(nv);
}

// scores[t,b] = dot(g_t[b,:], w_att)  -- cached once per step (deterministic)
__global__ void score_kernel(const float* __restrict__ g,
                             const float* __restrict__ w_att,
                             float* __restrict__ score_out) {
    __shared__ float red[256];
    int b = blockIdx.x, tid = threadIdx.x;
    float p = 0.0f;
    for (int d = tid; d < DHID; d += 256)
        p += g[((size_t)b << 10) + d] * w_att[d];
    red[tid] = p; __syncthreads();
    for (int s = 128; s > 0; s >>= 1) {
        if (tid < s) red[tid] += red[tid + s];
        __syncthreads();
    }
    if (tid == 0) score_out[b] = red[0];
}

// softmax over cached scores[0..t-1] + context; writes xp = [feat_t, c] bf16
__global__ void attn_kernel(const float* __restrict__ scores,   // T*B
                            const float* __restrict__ g_hist,   // T*B*D
                            const unsigned short* __restrict__ featT,
                            unsigned short* __restrict__ xp, int t) {
    int b = blockIdx.x, tid = threadIdx.x;
    for (int d = tid; d < DHID; d += 256)
        xp[(size_t)b * 2048 + d] = featT[((size_t)b << 10) + d];
    if (t == 0) {
        for (int d = tid; d < DHID; d += 256)
            xp[(size_t)b * 2048 + 1024 + d] = 0;   // empty history -> zero ctx
        return;
    }
    __shared__ float sh[256];
    __shared__ float alpha[T_STEPS];
    float v = (tid < t) ? scores[(size_t)tid * BATCH + b] : -3.0e38f;
    sh[tid] = v; __syncthreads();
    for (int s = 128; s > 0; s >>= 1) {
        if (tid < s) sh[tid] = fmaxf(sh[tid], sh[tid + s]);
        __syncthreads();
    }
    float mx = sh[0]; __syncthreads();
    float e = (tid < t) ? __expf(v - mx) : 0.0f;
    sh[tid] = e; __syncthreads();
    for (int s = 128; s > 0; s >>= 1) {
        if (tid < s) sh[tid] += sh[tid + s];
        __syncthreads();
    }
    float inv = 1.0f / sh[0];
    if (tid < t) alpha[tid] = e * inv;
    __syncthreads();
    for (int d = tid; d < DHID; d += 256) {
        float acc = 0.0f;
        for (int tt = 0; tt < t; ++tt)
            acc += alpha[tt] * g_hist[((size_t)tt * BATCH + b) * DHID + d];
        xp[(size_t)b * 2048 + 1024 + d] = f2bf(acc);
    }
}

// ---------------------------------------------------------------------------
// Host side
// ---------------------------------------------------------------------------
extern "C" void kernel_launch(void* const* d_in, const int* in_sizes, int n_in,
                              void* d_out, int out_size, void* d_ws, size_t ws_size,
                              hipStream_t stream) {
    const float* feat     = (const float*)d_in[0];   // T,B,DM
    const float* speakers = (const float*)d_in[1];   // T,B,S
    const float* Wih_g = (const float*)d_in[2];
    const float* Whh_g = (const float*)d_in[3];
    const float* bih_g = (const float*)d_in[4];
    const float* bhh_g = (const float*)d_in[5];
    const float* Wih_p = (const float*)d_in[6];
    const float* Whh_p = (const float*)d_in[7];
    const float* bih_p = (const float*)d_in[8];
    const float* bhh_p = (const float*)d_in[9];
    const float* Wih_e = (const float*)d_in[10];
    const float* Whh_e = (const float*)d_in[11];
    const float* bih_e = (const float*)d_in[12];
    const float* bhh_e = (const float*)d_in[13];
    const float* w_att = (const float*)d_in[14];
    float* out = (float*)d_out;                      // T,B,DE fp32

    const size_t NWB = (size_t)N3 * 2048;            // big weight elem count
    const size_t NWS = (size_t)N3 * 1024;            // small weight elem count
    const size_t NTB = (size_t)T_STEPS * BATCH * DHID;
    const size_t NBD = (size_t)BATCH * DHID;

    char* wp = (char*)d_ws;
    auto alloc = [&](size_t bytes) -> void* {
        void* p = (void*)wp;
        wp += (bytes + 15) & ~(size_t)15;
        return p;
    };
    unsigned short* wih_g_bf = (unsigned short*)alloc(NWB * 2);
    unsigned short* whh_g_bf = (unsigned short*)alloc(NWS * 2);
    unsigned short* wih_p_bf = (unsigned short*)alloc(NWB * 2);
    unsigned short* whh_p_bf = (unsigned short*)alloc(NWS * 2);
    unsigned short* wih_e_bf = (unsigned short*)alloc(NWS * 2);
    unsigned short* whh_e_bf = (unsigned short*)alloc(NWS * 2);
    unsigned short* feat_bf  = (unsigned short*)alloc(NTB * 2);
    float* g_hist  = (float*)alloc(NTB * 4);
    float* scores  = (float*)alloc((size_t)T_STEPS * BATCH * 4);
    float* parties = (float*)alloc((size_t)BATCH * SPK * DHID * 4);
    unsigned short* parties_bf = (unsigned short*)alloc((size_t)BATCH * SPK * DHID * 2);
    unsigned short* gprev_bf   = (unsigned short*)alloc(NBD * 2);
    unsigned short* eprev_bf   = (unsigned short*)alloc(NBD * 2);
    unsigned short* ein_bf     = (unsigned short*)alloc(NBD * 2);
    unsigned short* xg         = (unsigned short*)alloc((size_t)BATCH * 2048 * 2);
    unsigned short* xp         = (unsigned short*)alloc((size_t)BATCH * 2048 * 2);
    float* gi_g = (float*)alloc((size_t)BATCH * N3 * 4);
    float* gh_g = (float*)alloc((size_t)BATCH * N3 * 4);
    float* gi_p = (float*)alloc((size_t)BATCH * N3 * 4);
    float* gh_p = (float*)alloc((size_t)BATCH * SPK * N3 * 4);
    float* gi_e = (float*)alloc((size_t)BATCH * N3 * 4);
    float* gh_e = (float*)alloc((size_t)BATCH * N3 * 4);
    int*   spk  = (int*)alloc((size_t)T_STEPS * BATCH * 4);
    float* zero_f = (float*)alloc(NBD * 4);
    unsigned short* zero_bf = (unsigned short*)alloc(NBD * 2);

    // ---- one-time (per launch) conversions & state init (deterministic) ----
    auto cvt = [&](const float* s, unsigned short* d, size_t n) {
        cvt_f32_bf16_kernel<<<2048, 256, 0, stream>>>(s, d, n);
    };
    cvt(Wih_g, wih_g_bf, NWB);
    cvt(Whh_g, whh_g_bf, NWS);
    cvt(Wih_p, wih_p_bf, NWB);
    cvt(Whh_p, whh_p_bf, NWS);
    cvt(Wih_e, wih_e_bf, NWS);
    cvt(Whh_e, whh_e_bf, NWS);
    cvt(feat,  feat_bf,  NTB);
    spk_idx_kernel<<<(T_STEPS * BATCH + 255) / 256, 256, 0, stream>>>(
        speakers, spk, T_STEPS * BATCH);
    zero_u32_kernel<<<512, 256, 0, stream>>>((unsigned int*)parties,
                                             (size_t)BATCH * SPK * DHID);
    zero_u32_kernel<<<512, 256, 0, stream>>>((unsigned int*)parties_bf,
                                             (size_t)BATCH * SPK * DHID / 2);
    zero_u32_kernel<<<512, 256, 0, stream>>>((unsigned int*)zero_f, NBD);
    zero_u32_kernel<<<512, 256, 0, stream>>>((unsigned int*)zero_bf, NBD / 2);
    zero_u32_kernel<<<512, 256, 0, stream>>>((unsigned int*)gprev_bf, NBD / 2);
    zero_u32_kernel<<<512, 256, 0, stream>>>((unsigned int*)eprev_bf, NBD / 2);

    const dim3 grid_g(96, 1, 2);   // z0: gi (K=2048), z1: gh (M=64, K=1024)
    const dim3 grid_p(96, 1, 3);   // z0: gi (K=2048), z1-2: gh (M=128, K=1024)
    const dim3 grid_e(96, 1, 2);   // z0: gi (K=1024), z1: gh (M=64, K=1024)
    const int EB = (BATCH * DHID + 255) / 256;           // elementwise blocks
    const int PB = (BATCH * SPK * DHID + 255) / 256;

    for (int t = 0; t < T_STEPS; ++t) {
        const unsigned short* featT = feat_bf + (size_t)t * NBD;
        const int* spkT = spk + (size_t)t * BATCH;
        float* g_t = g_hist + (size_t)t * NBD;
        const float* g_prev_f = t ? (g_hist + (size_t)(t - 1) * NBD) : zero_f;
        const unsigned short* g_prev_b = t ? gprev_bf : zero_bf;
        const float* e_prev_f = t ? (out + (size_t)(t - 1) * NBD) : zero_f;
        const unsigned short* e_prev_b = t ? eprev_bf : zero_bf;

        // 1) global GRU (gi + gh in one launch)
        build_xg_kernel<<<EB, 256, 0, stream>>>(featT, parties, spkT, xg);
        gemm_dual_bf16_wmma_kernel<<<grid_g, 256, 0, stream>>>(
            xg, wih_g_bf, gi_g, 2048, g_prev_b, whh_g_bf, gh_g, 1024);
        gate_gru_kernel<<<EB, 256, 0, stream>>>(gi_g, gh_g, g_prev_f,
                                                bih_g, bhh_g, g_t, gprev_bf);
        // 2) attention over history (cached scores)
        score_kernel<<<BATCH, 256, 0, stream>>>(g_t, w_att, scores + (size_t)t * BATCH);
        attn_kernel<<<BATCH, 256, 0, stream>>>(scores, g_hist, featT, xp, t);
        // 3) party GRU (gi shared over parties; gh over B*S rows)
        gemm_dual_bf16_wmma_kernel<<<grid_p, 256, 0, stream>>>(
            xp, wih_p_bf, gi_p, 2048, parties_bf, whh_p_bf, gh_p, 1024);
        gate_party_kernel<<<PB, 256, 0, stream>>>(gi_p, gh_p, parties,
                                                  bih_p, bhh_p, spkT,
                                                  parties_bf, ein_bf);
        // 4) emotion GRU -> output[t]
        gemm_dual_bf16_wmma_kernel<<<grid_e, 256, 0, stream>>>(
            ein_bf, wih_e_bf, gi_e, 1024, e_prev_b, whh_e_bf, gh_e, 1024);
        gate_gru_kernel<<<EB, 256, 0, stream>>>(gi_e, gh_e, e_prev_f,
                                                bih_e, bhh_e,
                                                out + (size_t)t * NBD, eprev_bf);
    }
}
